// CombinedAttnProcessor_single_41412074668262
// MI455X (gfx1250) — compile-verified, matrix operand order checked by reading, not run on hardware
//
#include <hip/hip_runtime.h>
#include <hip/hip_bf16.h>
#include <math.h>
#include <stdint.h>

typedef __attribute__((ext_vector_type(16))) __bf16 v16bf;
typedef __attribute__((ext_vector_type(8)))  __bf16 bf16x8;
typedef __attribute__((ext_vector_type(8)))  float  v8f;
typedef __attribute__((ext_vector_type(4)))  float  f32x4;
typedef __attribute__((ext_vector_type(4)))  unsigned u32x4;
typedef __attribute__((ext_vector_type(8)))  int      i32x8;
typedef __attribute__((ext_vector_type(4)))  int      i32x4;

#define DIMC   3072
#define SEQ    3072
#define NHEADS 24
#define HD     128
#define RANKC  64
#define BLOCKQ 2048
#define CONDQ  1024
#define IPTOK  128

#if defined(__has_builtin)
#  if __has_builtin(__builtin_amdgcn_tensor_load_to_lds) && \
      __has_builtin(__builtin_amdgcn_s_wait_tensorcnt)
#    define HAVE_TDM 1
#  else
#    define HAVE_TDM 0
#  endif
#else
#  define HAVE_TDM 0
#endif

// ---------------------------------------------------------------------------
// 16x32 bf16 fragment loader (A or B operand of v_wmma_f32_16x16x32_bf16).
// CDNA5 layout (ISA 7.12.2): lane L covers row (L&15); lanes 0-15 hold K
// chunks {0..7, 16..23}, lanes 16-31 hold {8..15, 24..31}.
// ---------------------------------------------------------------------------
__device__ __forceinline__ v16bf load_frag(const __bf16* p, int ld) {
    const int lane = threadIdx.x & 31;
    const int r    = lane & 15;
    const int half = lane >> 4;
    const __bf16* row = p + (size_t)r * ld;
    bf16x8 lo = *(const bf16x8*)(row + half * 8);
    bf16x8 hi = *(const bf16x8*)(row + 16 + half * 8);
    v16bf out;
#pragma unroll
    for (int i = 0; i < 8; ++i) { out[i] = lo[i]; out[8 + i] = hi[i]; }
    return out;
}

__device__ __forceinline__ v8f wmma_bf16(v16bf a, v16bf b, v8f c) {
    return __builtin_amdgcn_wmma_f32_16x16x32_bf16(false, a, false, b,
                                                   (short)0, c, false, false);
}

#if HAVE_TDM
// ---------------------------------------------------------------------------
// TDM 2-D tile load: global (row-major, bf16, row stride `stride0` elements,
// tile = tile1 rows x tile0 elements) -> LDS at byte offset lds_addr.
// D# packing per CDNA5 ISA (count=1, type=2,
// data_size=2B, no multicast/pad/iterate). This toolchain exposes the
// 6-argument builtin (g0, g1, g2, g3, extra, cpol).
// ---------------------------------------------------------------------------
__device__ __forceinline__ void tdm_load_2d(unsigned lds_addr,
                                            const void* gaddr,
                                            unsigned tile0, unsigned tile1,
                                            unsigned stride0)
{
    const unsigned long long ga = (unsigned long long)(uintptr_t)gaddr;
    u32x4 g0;
    g0[0] = 1u;                                         // count=1 (user D#)
    g0[1] = lds_addr;                                   // lds_addr[31:0]
    g0[2] = (unsigned)(ga & 0xFFFFFFFFull);             // global_addr[31:0]
    g0[3] = (unsigned)((ga >> 32) & 0x01FFFFFFull)      // global_addr[56:32]
          | (2u << 30);                                 // type=2 ("image")
    i32x8 g1;
    g1[0] = (int)(1u << 16);                            // data_size=2 bytes
    g1[1] = (int)((tile0 & 0xFFFFu) << 16);             // tensor_dim0[15:0]
    g1[2] = (int)((tile0 >> 16) | ((tile1 & 0xFFFFu) << 16)); // dim0 hi | dim1 lo
    g1[3] = (int)((tile1 >> 16) | (tile0 << 16));       // dim1 hi | tile_dim0
    g1[4] = (int)(tile1 & 0xFFFFu);                     // tile_dim1 (tile_dim2=0)
    g1[5] = (int)stride0;                               // tensor_dim0_stride lo
    g1[6] = 0;                                          // stride0 hi | stride1 lo
    g1[7] = 0;
    i32x4 z4;
    z4[0] = 0; z4[1] = 0; z4[2] = 0; z4[3] = 0;
    i32x8 z8;
#pragma unroll
    for (int i = 0; i < 8; ++i) z8[i] = 0;
    __builtin_amdgcn_tensor_load_to_lds(g0, g1, z4, z4, z8, 0);
}
#endif

// ---------------------------------------------------------------------------
// Generic TN GEMM:  C[M,N] = alpha * A[M,K] * B[N,K]^T + bias[N] (+ C)
// A, B f32 row-major; converted to bf16 while staging into LDS.
// Tile 128x128x32, 256 threads (8 waves), each wave computes 32x64.
// ---------------------------------------------------------------------------
__global__ __launch_bounds__(256)
void gemm_tn_kernel(const float* __restrict__ A, int lda,
                    const float* __restrict__ B, int ldb,
                    float* __restrict__ C, int ldc,
                    int M, int N, int K,
                    const float* __restrict__ bias,
                    float alpha, int accumulate)
{
    __shared__ __align__(16) __bf16 As[128 * 32];
    __shared__ __align__(16) __bf16 Bs[128 * 32];

    const int tid   = threadIdx.x;
    const int tileM = blockIdx.y * 128;
    const int tileN = blockIdx.x * 128;
    const int wave  = tid >> 5;
    const int wm    = wave & 3;   // 4 M-groups of 32 rows
    const int wn    = wave >> 2;  // 2 N-groups of 64 cols

    v8f acc[2][4];
#pragma unroll
    for (int i = 0; i < 2; ++i)
#pragma unroll
        for (int j = 0; j < 4; ++j)
#pragma unroll
            for (int e = 0; e < 8; ++e) acc[i][j][e] = 0.0f;

    const int srow = tid >> 1;          // 0..127
    const int scol = (tid & 1) * 16;    // 0 or 16

    for (int k0 = 0; k0 < K; k0 += 32) {
        // stage A
        {
            const float* src = A + (size_t)(tileM + srow) * lda + k0 + scol;
            __bf16* dst = As + srow * 32 + scol;
#pragma unroll
            for (int i = 0; i < 4; ++i) {
                f32x4 v = *(const f32x4*)(src + i * 4);
#pragma unroll
                for (int c = 0; c < 4; ++c) dst[i * 4 + c] = (__bf16)v[c];
            }
            if (k0 + 32 < K) __builtin_prefetch(src + 32, 0, 1);
        }
        // stage B (guard N)
        {
            const int brow = tileN + srow;
            __bf16* dst = Bs + srow * 32 + scol;
            if (brow < N) {
                const float* src = B + (size_t)brow * ldb + k0 + scol;
#pragma unroll
                for (int i = 0; i < 4; ++i) {
                    f32x4 v = *(const f32x4*)(src + i * 4);
#pragma unroll
                    for (int c = 0; c < 4; ++c) dst[i * 4 + c] = (__bf16)v[c];
                }
                if (k0 + 32 < K) __builtin_prefetch(src + 32, 0, 1);
            } else {
#pragma unroll
                for (int i = 0; i < 16; ++i) dst[i] = (__bf16)0.0f;
            }
        }
        __syncthreads();

        v16bf af[2], bfm[4];
#pragma unroll
        for (int i = 0; i < 2; ++i)
            af[i] = load_frag(As + (wm * 32 + i * 16) * 32, 32);
#pragma unroll
        for (int j = 0; j < 4; ++j)
            bfm[j] = load_frag(Bs + (wn * 64 + j * 16) * 32, 32);
#pragma unroll
        for (int i = 0; i < 2; ++i)
#pragma unroll
            for (int j = 0; j < 4; ++j)
                acc[i][j] = wmma_bf16(af[i], bfm[j], acc[i][j]);
        __syncthreads();
    }

    const int lane = tid & 31;
    const int nidx = lane & 15;
    const int half = lane >> 4;
#pragma unroll
    for (int j = 0; j < 4; ++j) {
        const int col = tileN + wn * 64 + j * 16 + nidx;
        if (col >= N) continue;
        const float bv = bias ? bias[col] : 0.0f;
#pragma unroll
        for (int i = 0; i < 2; ++i) {
#pragma unroll
            for (int e = 0; e < 8; ++e) {
                const int row = tileM + wm * 32 + i * 16 + e + half * 8;
                float v = alpha * acc[i][j][e] + bv;
                size_t idx = (size_t)row * ldc + col;
                if (accumulate) v += C[idx];
                C[idx] = v;
            }
        }
    }
}

// ---------------------------------------------------------------------------
// RMSNorm (+ optional weight, + optional RoPE):
// src f32 [rows][DIMC] -> dst bf16 [NHEADS][rows][HD]
// ---------------------------------------------------------------------------
__global__ __launch_bounds__(128)
void norm_rope_kernel(const float* __restrict__ src,
                      const float* __restrict__ w,
                      const float* __restrict__ cs,
                      const float* __restrict__ sn,
                      __bf16* __restrict__ dst,
                      int rows, float eps)
{
    const int s = blockIdx.x;
    const int h = blockIdx.y;
    const int d = threadIdx.x;

    float x = src[(size_t)s * DIMC + h * HD + d];
    float ss = x * x;
#pragma unroll
    for (int off = 16; off > 0; off >>= 1) ss += __shfl_xor(ss, off, 32);

    __shared__ float red[4];
    if ((d & 31) == 0) red[d >> 5] = ss;
    __syncthreads();
    const float tot = red[0] + red[1] + red[2] + red[3];
    float y = x * rsqrtf(tot * (1.0f / HD) + eps);
    if (w) y *= w[d];
    if (cs) {
        const float yp  = __shfl_xor(y, 1, 32);
        const float rot = (d & 1) ? yp : -yp;
        y = y * cs[(size_t)s * HD + d] + rot * sn[(size_t)s * HD + d];
    }
    dst[((size_t)h * rows + s) * HD + d] = (__bf16)y;
}

// ---------------------------------------------------------------------------
// V transpose: src f32 [T][DIMC] -> dst bf16 [NHEADS][HD][T]
// ---------------------------------------------------------------------------
__global__ __launch_bounds__(256)
void transpose_v_kernel(const float* __restrict__ src,
                        __bf16* __restrict__ dst, int T)
{
    const size_t i = (size_t)blockIdx.x * 256 + threadIdx.x;
    const size_t total = (size_t)T * DIMC;
    if (i >= total) return;
    const int t = (int)(i / DIMC);
    const int c = (int)(i % DIMC);
    const int h = c / HD, d = c % HD;
    dst[((size_t)h * HD + d) * T + t] = (__bf16)src[i];
}

// ---------------------------------------------------------------------------
// Flash attention. Block = (head, 128 q rows), 8 waves, one 16-row q tile
// each. Per 32-key step, K tile (32x128) and V^T tile (128x32) are staged
// into LDS once per block (TDM async-tensor load when available) and shared
// by all 8 waves. Block mask is tile-granular and block-uniform.
// Q bf16 [H][SEQ][HD], K bf16 [H][T][HD], VT bf16 [H][HD][T], O f32 [SEQ][DIMC]
// ---------------------------------------------------------------------------
__global__ __launch_bounds__(256)
void flash_attn_kernel(const __bf16* __restrict__ Q,
                       const __bf16* __restrict__ Kb,
                       const __bf16* __restrict__ VT,
                       float* __restrict__ O,
                       int T, int use_mask, float sm_scale)
{
    __shared__ __align__(16) __bf16 Ktile[32 * HD];   // [t][d]  8KB
    __shared__ __align__(16) __bf16 Vtile[HD * 32];   // [d][t]  8KB
    __shared__ __align__(16) __bf16 Plds[8 * 16 * 32];

    const int qblocks = SEQ / 128;                // 24
    const int h   = blockIdx.x / qblocks;
    const int qb  = blockIdx.x % qblocks;
    const int wib = threadIdx.x >> 5;
    const int q0  = qb * 128 + wib * 16;
    const int lane = threadIdx.x & 31;
    const int nidx = lane & 15;
    const int half = lane >> 4;
    __bf16* Pw = Plds + wib * 16 * 32;

    const __bf16* qp = Q + ((size_t)h * SEQ + q0) * HD;
    v16bf aq[4];
#pragma unroll
    for (int kk = 0; kk < 4; ++kk) aq[kk] = load_frag(qp + kk * 32, HD);

    v8f acc[8];
    float mrow[8], lrow[8];
#pragma unroll
    for (int j = 0; j < 8; ++j)
#pragma unroll
        for (int e = 0; e < 8; ++e) acc[j][e] = 0.0f;
#pragma unroll
    for (int e = 0; e < 8; ++e) { mrow[e] = -1e30f; lrow[e] = 0.0f; }

    const int masked = use_mask && (qb * 128 >= BLOCKQ);   // block-uniform
    const int t0 = masked ? BLOCKQ : 0;
    const int t1 = masked ? (BLOCKQ + CONDQ) : T;

#if HAVE_TDM
    const unsigned kLds = (unsigned)(uintptr_t)Ktile;
    const unsigned vLds = (unsigned)(uintptr_t)Vtile;
#endif

    for (int kt = t0; kt < t1; kt += 32) {
        __syncthreads();   // previous-step LDS reads done before overwrite
#if HAVE_TDM
        if (wib == 0) {
            tdm_load_2d(kLds, Kb + ((size_t)h * T + kt) * HD, HD, 32, HD);
            tdm_load_2d(vLds, VT + (size_t)h * HD * T + kt, 32, HD, (unsigned)T);
            __builtin_amdgcn_s_wait_tensorcnt(0);
        }
#else
        {
            // cooperative staging: 256 threads x 16 bf16 per tile
            const int kr = threadIdx.x >> 3;             // 0..31 key row
            const int kc = (threadIdx.x & 7) * 16;       // 0..112
            const __bf16* ks = Kb + ((size_t)h * T + kt + kr) * HD + kc;
            *(bf16x8*)(Ktile + kr * HD + kc)     = *(const bf16x8*)(ks);
            *(bf16x8*)(Ktile + kr * HD + kc + 8) = *(const bf16x8*)(ks + 8);
            const int vr = threadIdx.x >> 1;             // 0..127 d row
            const int vc = (threadIdx.x & 1) * 16;       // 0 or 16
            const __bf16* vs = VT + ((size_t)h * HD + vr) * T + kt + vc;
            *(bf16x8*)(Vtile + vr * 32 + vc)     = *(const bf16x8*)(vs);
            *(bf16x8*)(Vtile + vr * 32 + vc + 8) = *(const bf16x8*)(vs + 8);
        }
#endif
        __syncthreads();   // tiles visible to all waves

        v8f s0, s1;
#pragma unroll
        for (int e = 0; e < 8; ++e) { s0[e] = 0.0f; s1[e] = 0.0f; }
#pragma unroll
        for (int kk = 0; kk < 4; ++kk) {
            v16bf b0 = load_frag(Ktile + kk * 32, HD);
            v16bf b1 = load_frag(Ktile + 16 * HD + kk * 32, HD);
            s0 = wmma_bf16(aq[kk], b0, s0);
            s1 = wmma_bf16(aq[kk], b1, s1);
        }

        float p0[8], p1[8];
#pragma unroll
        for (int e = 0; e < 8; ++e) {
            p0[e] = s0[e] * sm_scale;
            p1[e] = s1[e] * sm_scale;
        }
#pragma unroll
        for (int e = 0; e < 8; ++e) {
            float rm = fmaxf(p0[e], p1[e]);
            rm = fmaxf(rm, __shfl_xor(rm, 1, 32));
            rm = fmaxf(rm, __shfl_xor(rm, 2, 32));
            rm = fmaxf(rm, __shfl_xor(rm, 4, 32));
            rm = fmaxf(rm, __shfl_xor(rm, 8, 32));
            const float mnew = fmaxf(mrow[e], rm);
            const float sc   = __expf(mrow[e] - mnew);
            mrow[e] = mnew;
            p0[e] = __expf(p0[e] - mnew);
            p1[e] = __expf(p1[e] - mnew);
            float rs = p0[e] + p1[e];
            rs += __shfl_xor(rs, 1, 32);
            rs += __shfl_xor(rs, 2, 32);
            rs += __shfl_xor(rs, 4, 32);
            rs += __shfl_xor(rs, 8, 32);
            lrow[e] = lrow[e] * sc + rs;
#pragma unroll
            for (int j = 0; j < 8; ++j) acc[j][e] *= sc;
        }

        // P (16x32 bf16) -> wave-private LDS, reload as A fragment
#pragma unroll
        for (int e = 0; e < 8; ++e) {
            const int m = e + half * 8;
            Pw[m * 32 + nidx]      = (__bf16)p0[e];
            Pw[m * 32 + 16 + nidx] = (__bf16)p1[e];
        }
        asm volatile("s_wait_dscnt 0" ::: "memory");
        v16bf pa = load_frag(Pw, 32);

#pragma unroll
        for (int j = 0; j < 8; ++j) {
            v16bf bv = load_frag(Vtile + (j * 16) * 32, 32);
            acc[j] = wmma_bf16(pa, bv, acc[j]);
        }
    }

#pragma unroll
    for (int j = 0; j < 8; ++j) {
        const int col = h * HD + j * 16 + nidx;
#pragma unroll
        for (int e = 0; e < 8; ++e) {
            const int row = q0 + e + half * 8;
            O[(size_t)row * DIMC + col] = acc[j][e] / lrow[e];
        }
    }
}

__global__ __launch_bounds__(256)
void add_scaled_kernel(float* __restrict__ out, const float* __restrict__ a,
                       float scale, size_t n)
{
    const size_t i = (size_t)blockIdx.x * 256 + threadIdx.x;
    if (i < n) out[i] += scale * a[i];
}

// ---------------------------------------------------------------------------

extern "C" void kernel_launch(void* const* d_in, const int* in_sizes, int n_in,
                              void* d_out, int out_size, void* d_ws, size_t ws_size,
                              hipStream_t stream)
{
    const float* hs      = (const float*)d_in[0];
    const float* img     = (const float*)d_in[1];
    const float* wq      = (const float*)d_in[2];
    const float* bq      = (const float*)d_in[3];
    const float* wk      = (const float*)d_in[4];
    const float* bk      = (const float*)d_in[5];
    const float* wv      = (const float*)d_in[6];
    const float* bv      = (const float*)d_in[7];
    const float* lqd     = (const float*)d_in[8];
    const float* lqu     = (const float*)d_in[9];
    const float* lkd     = (const float*)d_in[10];
    const float* lku     = (const float*)d_in[11];
    const float* lvd     = (const float*)d_in[12];
    const float* lvu     = (const float*)d_in[13];
    const float* nqw     = (const float*)d_in[14];
    const float* nkw     = (const float*)d_in[15];
    const float* wkip    = (const float*)d_in[16];
    const float* wvip    = (const float*)d_in[17];
    const float* rcos    = (const float*)d_in[18];
    const float* rsin    = (const float*)d_in[19];
    float* out = (float*)d_out;

    char* base = (char*)d_ws;
    size_t off = 0;
    auto alloc = [&](size_t bytes) -> char* {
        char* p = base + off;
        off = (off + bytes + 255) & ~(size_t)255;
        return p;
    };
    float*  qf    = (float*)alloc((size_t)SEQ * DIMC * 4);
    float*  kf    = (float*)alloc((size_t)SEQ * DIMC * 4);
    float*  vf    = (float*)alloc((size_t)SEQ * DIMC * 4);
    float*  rq    = (float*)alloc((size_t)CONDQ * RANKC * 4);
    float*  rk    = (float*)alloc((size_t)CONDQ * RANKC * 4);
    float*  rv    = (float*)alloc((size_t)CONDQ * RANKC * 4);
    float*  kipf  = (float*)alloc((size_t)IPTOK * DIMC * 4);
    float*  vipf  = (float*)alloc((size_t)IPTOK * DIMC * 4);
    __bf16* qbf   = (__bf16*)alloc((size_t)NHEADS * SEQ * HD * 2);
    __bf16* kbf   = (__bf16*)alloc((size_t)NHEADS * SEQ * HD * 2);
    __bf16* vtbf  = (__bf16*)alloc((size_t)NHEADS * HD * SEQ * 2);
    __bf16* kipbf = (__bf16*)alloc((size_t)NHEADS * IPTOK * HD * 2);
    __bf16* viptb = (__bf16*)alloc((size_t)NHEADS * HD * IPTOK * 2);
    float*  ipbuf = (float*)alloc((size_t)SEQ * DIMC * 4);

    const float lscale   = 1.0f * (64.0f / (float)RANKC);  // LORA_W*(ALPHA/RANK)
    const float sm_scale = 0.088388347648318440550f;       // 1/sqrt(128)

    dim3 blk(256);
    auto gemm = [&](const float* A, int lda, const float* B, int ldb,
                    float* C, int ldc, int M, int N, int K,
                    const float* bias, float alpha, int accum) {
        dim3 grid((N + 127) / 128, M / 128);
        gemm_tn_kernel<<<grid, blk, 0, stream>>>(A, lda, B, ldb, C, ldc,
                                                 M, N, K, bias, alpha, accum);
    };

    // QKV projections (bias fused)
    gemm(hs, DIMC, wq, DIMC, qf, DIMC, SEQ, DIMC, DIMC, bq, 1.0f, 0);
    gemm(hs, DIMC, wk, DIMC, kf, DIMC, SEQ, DIMC, DIMC, bk, 1.0f, 0);
    gemm(hs, DIMC, wv, DIMC, vf, DIMC, SEQ, DIMC, DIMC, bv, 1.0f, 0);

    // LoRA (rows [BLOCKQ, BLOCKQ+CONDQ)): down, then up with accumulate
    const float* hsm = hs + (size_t)BLOCKQ * DIMC;
    gemm(hsm, DIMC, lqd, DIMC, rq, RANKC, CONDQ, RANKC, DIMC, nullptr, 1.0f, 0);
    gemm(hsm, DIMC, lkd, DIMC, rk, RANKC, CONDQ, RANKC, DIMC, nullptr, 1.0f, 0);
    gemm(hsm, DIMC, lvd, DIMC, rv, RANKC, CONDQ, RANKC, DIMC, nullptr, 1.0f, 0);
    gemm(rq, RANKC, lqu, RANKC, qf + (size_t)BLOCKQ * DIMC, DIMC,
         CONDQ, DIMC, RANKC, nullptr, lscale, 1);
    gemm(rk, RANKC, lku, RANKC, kf + (size_t)BLOCKQ * DIMC, DIMC,
         CONDQ, DIMC, RANKC, nullptr, lscale, 1);
    gemm(rv, RANKC, lvu, RANKC, vf + (size_t)BLOCKQ * DIMC, DIMC,
         CONDQ, DIMC, RANKC, nullptr, lscale, 1);

    // IP projections
    gemm(img, DIMC, wkip, DIMC, kipf, DIMC, IPTOK, DIMC, DIMC, nullptr, 1.0f, 0);
    gemm(img, DIMC, wvip, DIMC, vipf, DIMC, IPTOK, DIMC, DIMC, nullptr, 1.0f, 0);

    // epilogues
    norm_rope_kernel<<<dim3(SEQ, NHEADS), 128, 0, stream>>>(
        qf, nqw, rcos, rsin, qbf, SEQ, 1e-6f);
    norm_rope_kernel<<<dim3(SEQ, NHEADS), 128, 0, stream>>>(
        kf, nkw, rcos, rsin, kbf, SEQ, 1e-6f);
    norm_rope_kernel<<<dim3(IPTOK, NHEADS), 128, 0, stream>>>(
        kipf, nullptr, nullptr, nullptr, kipbf, IPTOK, 1e-5f);

    {
        size_t n = (size_t)SEQ * DIMC;
        transpose_v_kernel<<<(unsigned)((n + 255) / 256), blk, 0, stream>>>(vf, vtbf, SEQ);
        size_t m = (size_t)IPTOK * DIMC;
        transpose_v_kernel<<<(unsigned)((m + 255) / 256), blk, 0, stream>>>(vipf, viptb, IPTOK);
    }

    // attention: masked self-attn -> out; IP attn -> ipbuf; then add
    const int nblocks = NHEADS * (SEQ / 128);   // 576 blocks, 8 waves each
    flash_attn_kernel<<<nblocks, blk, 0, stream>>>(qbf, kbf, vtbf, out,
                                                   SEQ, 1, sm_scale);
    flash_attn_kernel<<<nblocks, blk, 0, stream>>>(qbf, kipbf, viptb, ipbuf,
                                                   IPTOK, 0, sm_scale);
    {
        size_t n = (size_t)SEQ * DIMC;
        add_scaled_kernel<<<(unsigned)((n + 255) / 256), blk, 0, stream>>>(
            out, ipbuf, 1.0f /*IP_SCALE*/, n);
    }
    (void)in_sizes; (void)n_in; (void)out_size; (void)ws_size;
}
